// SimpleRNN_31181462569443
// MI455X (gfx1250) — matrix operand order repaired and checked by reading, not compiled, not run on hardware
//
#include <hip/hip_runtime.h>
#include <math.h>

#define B_  32
#define S_  128
#define C_  16
#define E_  256
#define H_  512      // word hidden
#define NT_ 4096     // B*S
#define G_  2048     // 4*H word gates (per direction)
#define HC_ 1024     // char hidden
#define GC_ 4096     // 4*HC char gates
#define NC_ 20

// GEMM tiling
#define BM_ 128
#define BN_ 128
#define KC_ 64
#define LP_ 72       // padded LDS row pitch in halves (144B: 16B-aligned, conflict-free)

#if defined(__gfx1250__) && __has_builtin(__builtin_amdgcn_global_load_async_to_lds_b128)
#define HAVE_ASYNC 1
#else
#define HAVE_ASYNC 0
#endif

typedef __attribute__((ext_vector_type(16))) __bf16 v16bf;
typedef __attribute__((ext_vector_type(8)))  float  v8f;
typedef int v4i_ __attribute__((vector_size(16)));   // matches builtin's pointee type

union frag_u { v16bf v; uint4 q[2]; };

__device__ __forceinline__ unsigned short f2bf(float f) {
  unsigned u = __float_as_uint(f);
  unsigned r = 0x7FFFu + ((u >> 16) & 1u);   // round-to-nearest-even
  return (unsigned short)((u + r) >> 16);
}
__device__ __forceinline__ float sigmoidf_(float x) { return 1.0f / (1.0f + expf(-x)); }

// global(16B) -> LDS(16B) copy; async-to-LDS on gfx1250 when available
__device__ __forceinline__ void cp_g2s_b128(const unsigned short* g, unsigned short* s) {
#if HAVE_ASYNC
  __builtin_amdgcn_global_load_async_to_lds_b128(
      (__attribute__((address_space(1))) v4i_*)g,
      (__attribute__((address_space(3))) v4i_*)s, 0, 0);
#else
  *(uint4*)s = *(const uint4*)g;
#endif
}
__device__ __forceinline__ void cp_g2s_drain() {
#if HAVE_ASYNC
  asm volatile("s_wait_asynccnt 0" ::: "memory");
#endif
}

// ---------------- conversion / embedding ----------------
__global__ void k_f32_to_bf16(const float* __restrict__ in, unsigned short* __restrict__ out, long n) {
  long i = (long)blockIdx.x * blockDim.x + threadIdx.x;
  if (i < n) out[i] = f2bf(in[i]);
}

__global__ void k_embed_bf16(const int* __restrict__ ids, const float* __restrict__ tab,
                             unsigned short* __restrict__ out, long ntok, int E) {
  long i = (long)blockIdx.x * blockDim.x + threadIdx.x;
  if (i >= ntok * (long)E) return;
  long t = i / E; int e = (int)(i - t * E);
  out[i] = f2bf(tab[(size_t)ids[t] * E + e]);
}

// ---- WMMA bf16 GEMM, LDS-staged: D[M,N] = A[M,K] * W[N,K]^T (+bias) (+=) ----
// Block = 256 threads = 8 waves; block tile 128x128; wave tile 32(M) x 64(N).
// Requires: M % 16 == 0, N % 128 == 0, K % 64 == 0, 16B-aligned A/B rows.
__global__ __launch_bounds__(256) void k_gemm_bf16_wmma(
    const unsigned short* __restrict__ A, int lda,
    const unsigned short* __restrict__ Bw, int ldb,
    float* __restrict__ D, int ldd,
    const float* __restrict__ bias, int accumulate,
    int M, int N, int K)
{
  __shared__ __align__(16) unsigned short lA[BM_ * LP_];
  __shared__ __align__(16) unsigned short lB[BN_ * LP_];

  const int tid  = threadIdx.x;
  const int lane = tid & 31;
  const int wid  = tid >> 5;
  const int wm   = wid & 3;          // 0..3 -> 32-row M subtile
  const int wn   = wid >> 2;         // 0..1 -> 64-col N subtile
  const int mBlk = blockIdx.y * BM_;
  const int nBlk = blockIdx.x * BN_;

  const int r16 = lane & 15;
  const int hi  = lane >> 4;         // 0 or 1
  const int ka  = hi << 3;           // A K-sub-offset (ISA 16-bit A layout)
  const int kb  = hi << 3;           // B K-sub-offset (ISA 16-bit B layout)
  const int rb  = hi << 3;           // C/D row-sub-offset

  const int lrow = tid >> 3;         // 0..31 : staging row within 32-row group
  const int lcol = (tid & 7) << 3;   // 0,8,...,56 : staging col (halves)

  v8f zero = {};
  v8f acc[2][4];
#pragma unroll
  for (int i = 0; i < 2; ++i)
#pragma unroll
    for (int j = 0; j < 4; ++j) acc[i][j] = zero;

  for (int k0 = 0; k0 < K; k0 += KC_) {
    __syncthreads();
#pragma unroll
    for (int r = 0; r < 4; ++r) {
      int row = r * 32 + lrow;
      int ga = mBlk + row; if (ga > M - 1) ga = M - 1;   // clamp (store is guarded)
      int gb = nBlk + row; if (gb > N - 1) gb = N - 1;
      cp_g2s_b128(A  + (size_t)ga * lda + k0 + lcol, &lA[row * LP_ + lcol]);
      cp_g2s_b128(Bw + (size_t)gb * ldb + k0 + lcol, &lB[row * LP_ + lcol]);
    }
    cp_g2s_drain();
    __syncthreads();

#pragma unroll
    for (int kk = 0; kk < KC_; kk += 32) {
      frag_u af[2], bfr[4];
#pragma unroll
      for (int i = 0; i < 2; ++i) {
        const unsigned short* p = &lA[(wm * 32 + i * 16 + r16) * LP_ + kk + ka];
        af[i].q[0] = *(const uint4*)p;
        af[i].q[1] = *(const uint4*)(p + 16);
      }
#pragma unroll
      for (int j = 0; j < 4; ++j) {
        const unsigned short* p = &lB[(wn * 64 + j * 16 + r16) * LP_ + kk + kb];
        bfr[j].q[0] = *(const uint4*)p;
        bfr[j].q[1] = *(const uint4*)(p + 16);
      }
#pragma unroll
      for (int i = 0; i < 2; ++i)
#pragma unroll
        for (int j = 0; j < 4; ++j)
          acc[i][j] = __builtin_amdgcn_wmma_f32_16x16x32_bf16(
              false, af[i].v, false, bfr[j].v, (short)0, acc[i][j], false, false);
    }
  }

#pragma unroll
  for (int i = 0; i < 2; ++i) {
    const int m0 = mBlk + wm * 32 + i * 16;
    if (m0 >= M) continue;
#pragma unroll
    for (int j = 0; j < 4; ++j) {
      const int col = nBlk + wn * 64 + j * 16 + r16;
      const float bv = bias ? bias[col] : 0.0f;
#pragma unroll
      for (int r = 0; r < 8; ++r) {
        size_t off = (size_t)(m0 + rb + r) * ldd + col;
        float v = acc[i][j][r] + bv;
        if (accumulate) D[off] += v; else D[off] = v;
      }
    }
  }
}

// ---------------- word LSTM cell (one timestep, one direction) ----------------
__global__ void k_lstm_cell_word(
    const float* __restrict__ gsum,
    const float* __restrict__ xg, int xg_col, int xg_ld,
    const int* __restrict__ lens, int t, int rev,
    float* __restrict__ c, unsigned short* __restrict__ hbf,
    float* __restrict__ y, int y_ld, int y_col)
{
  int i = blockIdx.x * blockDim.x + threadIdx.x;
  if (i >= B_ * H_) return;
  int n = i / H_, j = i - n * H_;
  int len = lens[n];
  if (t >= len) return;                      // masked step: h,c frozen, y stays 0
  int pos = rev ? (len - 1 - t) : t;
  size_t xr = (size_t)(n * S_ + pos) * xg_ld + xg_col;
  size_t gr = (size_t)n * G_;
  float gi = gsum[gr + j]          + xg[xr + j];
  float gf = gsum[gr + H_ + j]     + xg[xr + H_ + j];
  float gg = gsum[gr + 2 * H_ + j] + xg[xr + 2 * H_ + j];
  float go = gsum[gr + 3 * H_ + j] + xg[xr + 3 * H_ + j];
  float c2 = sigmoidf_(gf) * c[i] + sigmoidf_(gi) * tanhf(gg);
  float h2 = sigmoidf_(go) * tanhf(c2);
  c[i] = c2;
  hbf[i] = f2bf(h2);
  y[(size_t)(n * S_ + pos) * y_ld + y_col + j] = h2;
}

// ---------------- char LSTM cell (collect=False) ----------------
__global__ void k_lstm_cell_char(
    const float* __restrict__ g, const int* __restrict__ lens, int t,
    float* __restrict__ h, float* __restrict__ c, unsigned short* __restrict__ hbf)
{
  long i = (long)blockIdx.x * blockDim.x + threadIdx.x;
  if (i >= (long)NT_ * HC_) return;
  int n = (int)(i / HC_), j = (int)(i - (long)n * HC_);
  if (t >= lens[n]) return;
  size_t gr = (size_t)n * GC_;
  float gi = g[gr + j];
  float gf = g[gr + HC_ + j];
  float gg = g[gr + 2 * HC_ + j];
  float go = g[gr + 3 * HC_ + j];
  float c2 = sigmoidf_(gf) * c[i] + sigmoidf_(gi) * tanhf(gg);
  float h2 = sigmoidf_(go) * tanhf(c2);
  c[i] = c2; h[i] = h2; hbf[i] = f2bf(h2);
}

// ---------------- final: out = (y1 + mask*hT) @ Wlin^T + blin ----------------
__global__ void k_final_linear(
    const float* __restrict__ y1, const float* __restrict__ hT,
    const int* __restrict__ wlens,
    const float* __restrict__ Wlin, const float* __restrict__ blin,
    float* __restrict__ out)
{
  int i = blockIdx.x * blockDim.x + threadIdx.x;
  if (i >= NT_ * NC_) return;
  int row = i / NC_, k = i - row * NC_;
  int b = row / S_, s = row - b * S_;
  bool act = s < wlens[b];
  const float* xr = y1   + (size_t)row * (2 * H_);
  const float* hr = hT   + (size_t)row * (2 * H_);
  const float* wr = Wlin + (size_t)k   * (2 * H_);
  float acc = blin[k];
  for (int e = 0; e < 2 * H_; ++e)
    acc += (xr[e] + (act ? hr[e] : 0.0f)) * wr[e];
  out[i] = acc;
}

extern "C" void kernel_launch(void* const* d_in, const int* in_sizes, int n_in,
                              void* d_out, int out_size, void* d_ws, size_t ws_size,
                              hipStream_t stream)
{
  (void)in_sizes; (void)n_in; (void)out_size; (void)ws_size;
  const int*   words     = (const int*)d_in[0];
  const int*   words_num = (const int*)d_in[1];
  const int*   chars     = (const int*)d_in[2];
  const int*   char_num  = (const int*)d_in[3];
  const float* Wemb = (const float*)d_in[4];
  const float* Cemb = (const float*)d_in[5];
  const float* Wih0 = (const float*)d_in[6];
  const float* Whh0 = (const float*)d_in[7];
  const float* b0   = (const float*)d_in[8];
  const float* Wih1 = (const float*)d_in[9];
  const float* Whh1 = (const float*)d_in[10];
  const float* b1   = (const float*)d_in[11];
  const float* cWih = (const float*)d_in[12];
  const float* cWhh = (const float*)d_in[13];
  const float* cb   = (const float*)d_in[14];
  const float* Wlin = (const float*)d_in[15];
  const float* blin = (const float*)d_in[16];
  float* out = (float*)d_out;

  char* w = (char*)d_ws;
  auto alloc = [&](size_t bytes) -> void* {
    void* p = (void*)w;
    w += (bytes + 255) & ~(size_t)255;
    return p;
  };
  // bf16 weight copies
  unsigned short* wih0b = (unsigned short*)alloc((size_t)2 * G_ * E_      * 2);
  unsigned short* whh0b = (unsigned short*)alloc((size_t)2 * G_ * H_      * 2);
  unsigned short* wih1b = (unsigned short*)alloc((size_t)2 * G_ * 2 * H_  * 2);
  unsigned short* whh1b = (unsigned short*)alloc((size_t)2 * G_ * H_      * 2);
  unsigned short* cwihb = (unsigned short*)alloc((size_t)GC_ * E_         * 2);
  unsigned short* cwhhb = (unsigned short*)alloc((size_t)GC_ * HC_        * 2);
  // bf16 activations
  unsigned short* x0b  = (unsigned short*)alloc((size_t)NT_ * E_          * 2);
  unsigned short* y0b  = (unsigned short*)alloc((size_t)NT_ * 2 * H_      * 2);
  unsigned short* ceb  = (unsigned short*)alloc((size_t)NT_ * C_ * E_     * 2);
  unsigned short* hbfw = (unsigned short*)alloc((size_t)B_ * H_           * 2);
  unsigned short* hcb  = (unsigned short*)alloc((size_t)NT_ * HC_         * 2);
  // f32 buffers
  float* xg  = (float*)alloc((size_t)NT_ * GC_ * 4);     // x-gates; reused as char gates
  float* y0f = (float*)alloc((size_t)NT_ * 2 * H_ * 4);
  float* y1f = (float*)alloc((size_t)NT_ * 2 * H_ * 4);
  float* gs  = (float*)alloc((size_t)B_ * G_ * 4);
  float* cw  = (float*)alloc((size_t)B_ * H_ * 4);
  float* hc  = (float*)alloc((size_t)NT_ * HC_ * 4);
  float* cc  = (float*)alloc((size_t)NT_ * HC_ * 4);

  auto cvt = [&](const float* src, unsigned short* dst, long n) {
    k_f32_to_bf16<<<(unsigned)((n + 255) / 256), 256, 0, stream>>>(src, dst, n);
  };
  auto gemm = [&](const unsigned short* A, int lda, const unsigned short* Bw, int ldb,
                  float* D, int ldd, const float* bias, int acc, int M, int N, int K) {
    dim3 grid((N + BN_ - 1) / BN_, (M + BM_ - 1) / BM_);
    k_gemm_bf16_wmma<<<grid, 256, 0, stream>>>(A, lda, Bw, ldb, D, ldd, bias, acc, M, N, K);
  };

  // ---- weight conversion + embeddings ----
  cvt(Wih0, wih0b, 2L * G_ * E_);
  cvt(Whh0, whh0b, 2L * G_ * H_);
  cvt(Wih1, wih1b, 2L * G_ * 2 * H_);
  cvt(Whh1, whh1b, 2L * G_ * H_);
  cvt(cWih, cwihb, (long)GC_ * E_);
  cvt(cWhh, cwhhb, (long)GC_ * HC_);
  k_embed_bf16<<<(NT_ * E_ + 255) / 256, 256, 0, stream>>>(words, Wemb, x0b, NT_, E_);
  k_embed_bf16<<<(unsigned)(((long)NT_ * C_ * E_ + 255) / 256), 256, 0, stream>>>(
      chars, Cemb, ceb, (long)NT_ * C_, E_);

  // ---- one BiLSTM layer: xg for both dirs in one GEMM, then sequential steps ----
  auto run_bilstm = [&](const unsigned short* xin, int Kin,
                        const unsigned short* wih, const unsigned short* whh2,
                        const float* bias, float* yout) {
    gemm(xin, Kin, wih, Kin, xg, 2 * G_, bias, 0, NT_, 2 * G_, Kin);
    (void)hipMemsetAsync(yout, 0, (size_t)NT_ * 2 * H_ * sizeof(float), stream);
    for (int dir = 0; dir < 2; ++dir) {
      (void)hipMemsetAsync(hbfw, 0, (size_t)B_ * H_ * 2, stream);
      (void)hipMemsetAsync(cw,   0, (size_t)B_ * H_ * 4, stream);
      const unsigned short* whh = whh2 + (size_t)dir * G_ * H_;
      for (int t = 0; t < S_; ++t) {
        gemm(hbfw, H_, whh, H_, gs, G_, nullptr, 0, B_, G_, H_);
        k_lstm_cell_word<<<(B_ * H_ + 255) / 256, 256, 0, stream>>>(
            gs, xg, dir * G_, 2 * G_, words_num, t, dir, cw, hbfw, yout, 2 * H_, dir * H_);
      }
    }
  };

  run_bilstm(x0b, E_, wih0b, whh0b, b0, y0f);
  cvt(y0f, y0b, (long)NT_ * 2 * H_);
  run_bilstm(y0b, 2 * H_, wih1b, whh1b, b1, y1f);

  // ---- char LSTM (collect=False, keep hT) ----
  (void)hipMemsetAsync(hc,  0, (size_t)NT_ * HC_ * 4, stream);
  (void)hipMemsetAsync(cc,  0, (size_t)NT_ * HC_ * 4, stream);
  (void)hipMemsetAsync(hcb, 0, (size_t)NT_ * HC_ * 2, stream);
  for (int t = 0; t < C_; ++t) {
    gemm(ceb + (size_t)t * E_, C_ * E_, cwihb, E_, xg, GC_, cb, 0, NT_, GC_, E_);
    gemm(hcb, HC_, cwhhb, HC_, xg, GC_, nullptr, 1, NT_, GC_, HC_);
    long n = (long)NT_ * HC_;
    k_lstm_cell_char<<<(unsigned)((n + 255) / 256), 256, 0, stream>>>(xg, char_num, t, hc, cc, hcb);
  }

  // ---- final linear ----
  k_final_linear<<<(NT_ * NC_ + 255) / 256, 256, 0, stream>>>(y1f, hc, words_num, Wlin, blin, out);
}
